// MultiHeadAttention_67216238182520
// MI455X (gfx1250) — matrix operand order repaired and checked by reading, not compiled, not run on hardware
//
#include <hip/hip_runtime.h>
#include <hip/hip_bf16.h>
#include <stdint.h>

typedef __bf16 bf16;
typedef __attribute__((ext_vector_type(16))) __bf16 v16bf;
typedef __attribute__((ext_vector_type(8)))  __bf16 v8bf;
typedef __attribute__((ext_vector_type(4)))  __bf16 v4bf;
typedef __attribute__((ext_vector_type(8)))  float  v8f;
typedef int v4i_ __attribute__((vector_size(16)));  // matches async builtin param type

#define DMODEL 1024
#define NHEADS 16
#define DK     64
#define SEQL   2048
#define BATCH  4

#define BM 128
#define BN 128
#define BK 32
#define KP 40   // LDS pitch (bf16 elems) for BK=32 tiles: 80B rows, 16B aligned
#define TP 80   // LDS pitch (bf16 elems) for 64-wide tiles: 160B rows
#define NJB (SEQL / 64)

#if __has_builtin(__builtin_amdgcn_global_load_async_to_lds_b128) && \
    __has_builtin(__builtin_amdgcn_s_wait_asynccnt)
#define HAVE_ASYNC_LDS 1
#endif
#if __has_builtin(__builtin_amdgcn_update_dpp)
#define HAVE_DPP 1
#endif

#ifdef HAVE_ASYNC_LDS
// per-lane 16B DMA: LDS[lp] = MEM[gp], tracked by ASYNCcnt
#define ASYNC_CP16(gp, lp)                                  \
  __builtin_amdgcn_global_load_async_to_lds_b128(           \
      (v4i_*)(const void*)(gp), (v4i_*)(void*)(lp), 0, 0)
#else
#define ASYNC_CP16(gp, lp) (*(v8bf*)(lp) = *(const v8bf*)(gp))
#endif

static __device__ __forceinline__ v16bf cat8(v8bf lo, v8bf hi) {
  return __builtin_shufflevector(lo, hi, 0,1,2,3,4,5,6,7,8,9,10,11,12,13,14,15);
}

static __device__ __forceinline__ v8f wmma_bf16(v16bf a, v16bf b, v8f c) {
  return __builtin_amdgcn_wmma_f32_16x16x32_bf16(false, a, false, b, (short)0, c,
                                                 false, false);
}

// ---- 16-lane row reductions (stay inside each DPP16 row on wave32) ----------
#ifdef HAVE_DPP
template <int CTRL>
static __device__ __forceinline__ float dpp_mov(float x) {
  return __int_as_float(
      __builtin_amdgcn_update_dpp(0, __float_as_int(x), CTRL, 0xF, 0xF, true));
}
// xor1 (quad_perm 1,0,3,2) -> xor2 (quad_perm 2,3,0,1) -> ^7 (row_half_mirror)
// -> ^15 (row_mirror): 4 steps reduce all 16 lanes of the row.
static __device__ __forceinline__ float rowred_max(float x) {
  x = fmaxf(x, dpp_mov<0xB1>(x));
  x = fmaxf(x, dpp_mov<0x4E>(x));
  x = fmaxf(x, dpp_mov<0x141>(x));
  x = fmaxf(x, dpp_mov<0x140>(x));
  return x;
}
static __device__ __forceinline__ float rowred_sum(float x) {
  x += dpp_mov<0xB1>(x);
  x += dpp_mov<0x4E>(x);
  x += dpp_mov<0x141>(x);
  x += dpp_mov<0x140>(x);
  return x;
}
#else
static __device__ __forceinline__ float rowred_max(float x) {
#pragma unroll
  for (int off = 1; off < 16; off <<= 1) x = fmaxf(x, __shfl_xor(x, off, 32));
  return x;
}
static __device__ __forceinline__ float rowred_sum(float x) {
#pragma unroll
  for (int off = 1; off < 16; off <<= 1) x += __shfl_xor(x, off, 32);
  return x;
}
#endif

// ---------------------------------------------------------------------------
// GEMM: out[m,n] = sum_k A[m,k] * W[n,k] + bias[n]   (M = BATCH*SEQL, N=K=1024)
// MODE 0: A f32 -> bf16 head-split [B,H,S,Dk]          (Q, K projections)
// MODE 2: A f32 -> bf16 head-split transposed [B,H,Dk,S] (V projection;
//         rows g are consecutive s -> packed 16B epilogue stores)
// MODE 1: A bf16 (async-staged) -> f32 row-major       (output projection)
// ---------------------------------------------------------------------------
template <int MODE>
__global__ __launch_bounds__(256)
void gemm_dmodel(const void* __restrict__ Aptr, const float* __restrict__ W,
                 const float* __restrict__ bias, void* __restrict__ Out) {
  __shared__ bf16 As[BM * KP];
  __shared__ bf16 Bs[BN * KP];

  const int tid  = threadIdx.x;
  const int lane = tid & 31;
  const int wave = tid >> 5;
  const int wr   = wave & 1;
  const int wc   = wave >> 1;
  const int lh   = lane >> 4;
  const int ln   = lane & 15;
  const int m0   = blockIdx.y * BM;
  const int n0   = blockIdx.x * BN;

  v8f acc[4][2];
#pragma unroll
  for (int r = 0; r < 4; ++r)
#pragma unroll
    for (int c = 0; c < 2; ++c) acc[r][c] = {};

  for (int k0 = 0; k0 < DMODEL; k0 += BK) {
    __syncthreads();
    // prefetch next W k-slice into L2 while this tile is consumed
    if (k0 + BK < DMODEL)
      __builtin_prefetch(W + (size_t)(n0 + (tid >> 3)) * DMODEL + k0 + BK +
                             (tid & 7) * 4, 0, 1);
    // ---- stage A tile ----
    if (MODE != 1) {
      const float* A = (const float*)Aptr;
#pragma unroll
      for (int p = 0; p < 4; ++p) {
        const int r = (tid >> 3) + p * 32;
        const int c = (tid & 7) * 4;
        const float4 f = *(const float4*)(A + (size_t)(m0 + r) * DMODEL + k0 + c);
        v4bf v; v[0] = (bf16)f.x; v[1] = (bf16)f.y; v[2] = (bf16)f.z; v[3] = (bf16)f.w;
        *(v4bf*)(As + r * KP + c) = v;
      }
    } else {
      const bf16* A = (const bf16*)Aptr;
#pragma unroll
      for (int p = 0; p < 2; ++p) {
        const int r = (tid >> 2) + p * 64;
        const int c = (tid & 3) * 8;
        ASYNC_CP16(A + (size_t)(m0 + r) * DMODEL + k0 + c, As + r * KP + c);
      }
    }
    // ---- stage W tile (rows = output features n), f32 -> bf16 ----
#pragma unroll
    for (int p = 0; p < 4; ++p) {
      const int r = (tid >> 3) + p * 32;
      const int c = (tid & 7) * 4;
      const float4 f = *(const float4*)(W + (size_t)(n0 + r) * DMODEL + k0 + c);
      v4bf v; v[0] = (bf16)f.x; v[1] = (bf16)f.y; v[2] = (bf16)f.z; v[3] = (bf16)f.w;
      *(v4bf*)(Bs + r * KP + c) = v;
    }
#ifdef HAVE_ASYNC_LDS
    if (MODE == 1) __builtin_amdgcn_s_wait_asynccnt(0);
#endif
    __syncthreads();

    // ---- B fragments: lane=n, elems i<8: K=8h+i ; i>=8: K=16+8h+(i-8) ----
    v16bf bfrag[2];
#pragma unroll
    for (int c = 0; c < 2; ++c) {
      const bf16* p = Bs + (wc * 32 + c * 16 + ln) * KP + lh * 8;
      bfrag[c] = cat8(*(const v8bf*)p, *(const v8bf*)(p + 16));
    }
    // ---- A fragments: lane=m, elems i<8: K=8h+i ; i>=8: K=16+8h+(i-8) ----
#pragma unroll
    for (int r = 0; r < 4; ++r) {
      const bf16* p = As + (wr * 64 + r * 16 + ln) * KP + lh * 8;
      const v16bf afrag = cat8(*(const v8bf*)p, *(const v8bf*)(p + 16));
#pragma unroll
      for (int c = 0; c < 2; ++c)
        acc[r][c] = wmma_bf16(afrag, bfrag[c], acc[r][c]);
    }
  }

  float bcol[2];
#pragma unroll
  for (int c = 0; c < 2; ++c) bcol[c] = bias[n0 + wc * 32 + c * 16 + ln];

#pragma unroll
  for (int r = 0; r < 4; ++r)
#pragma unroll
    for (int c = 0; c < 2; ++c) {
      const int col = n0 + wc * 32 + c * 16 + ln;
      const int hh = col >> 6, dk = col & (DK - 1);
      if (MODE == 2) {
        // rows g+8*lh are consecutive s -> one packed 16B store per fragment
        const int row0 = m0 + wr * 64 + r * 16 + 8 * lh;
        const int bb = row0 >> 11, s0 = row0 & (SEQL - 1);
        v8bf pk;
#pragma unroll
        for (int g = 0; g < 8; ++g) pk[g] = (bf16)(acc[r][c][g] + bcol[c]);
        bf16* O = (bf16*)Out;
        *(v8bf*)(O + ((size_t)(bb * NHEADS + hh) * DK + dk) * SEQL + s0) = pk;
      } else {
#pragma unroll
        for (int g = 0; g < 8; ++g) {
          const int row = m0 + wr * 64 + r * 16 + g + 8 * lh;
          const float v = acc[r][c][g] + bcol[c];
          if (MODE == 0) {
            bf16* O = (bf16*)Out;
            const int bb = row >> 11, s = row & (SEQL - 1);
            O[(((size_t)(bb * NHEADS + hh) * SEQL + s) * DK) + dk] = (bf16)v;
          } else {
            float* O = (float*)Out;
            O[(size_t)row * DMODEL + col] = v;
          }
        }
      }
    }
}

// ---------------------------------------------------------------------------
// Fused flash attention; K in [B,H,S,Dk], V pre-transposed in [B,H,Dk,S].
// Grid: (SEQL/64, NHEADS, BATCH); 128 threads = 4 waves, 16 query rows each.
// Double-buffered async DMA staging of K/V tiles.
// ---------------------------------------------------------------------------
__global__ __launch_bounds__(128)
void attn_fused(const bf16* __restrict__ qh, const bf16* __restrict__ kh,
                const bf16* __restrict__ vh, bf16* __restrict__ ctx) {
  __shared__ bf16 kt[2][64 * TP];     // K tile [key][d]
  __shared__ bf16 vt[2][64 * TP];     // V tile [d][key] (already transposed)
  __shared__ bf16 pt[4][16 * TP];     // per-wave P staging [qrow][key]

  const int tid  = threadIdx.x;
  const int lane = tid & 31;
  const int wave = tid >> 5;
  const int lh   = lane >> 4;
  const int ln   = lane & 15;
  const int h    = blockIdx.y, b = blockIdx.z;
  const size_t base = (size_t)(b * NHEADS + h) * SEQL * DK;  // same for [S,Dk] and [Dk,S]
  const int q0 = blockIdx.x * 64 + wave * 16;

  // Q A-fragments, kept in registers for all key blocks
  v16bf qf[2];
#pragma unroll
  for (int ks = 0; ks < 2; ++ks) {
    const bf16* p = qh + base + (size_t)(q0 + ln) * DK + ks * 32 + lh * 8;
    qf[ks] = cat8(*(const v8bf*)p, *(const v8bf*)(p + 16));
  }

  v8f o[4];
#pragma unroll
  for (int c = 0; c < 4; ++c) o[c] = {};
  float mrow[8], lrow[8];
#pragma unroll
  for (int g = 0; g < 8; ++g) { mrow[g] = -3.0e38f; lrow[g] = 0.0f; }
  const float scale = 0.125f;  // 1/sqrt(64)

  // 8 async B128 copies per wave-lane per tile (4 K rows + 4 V rows)
  auto stage = [&](int jb, int buf) {
#pragma unroll
    for (int p4 = 0; p4 < 4; ++p4) {
      const int r = (tid >> 3) + p4 * 16;
      const int c = (tid & 7) * 8;
      ASYNC_CP16(kh + base + (size_t)(jb * 64 + r) * DK + c, &kt[buf][r * TP + c]);
      ASYNC_CP16(vh + base + (size_t)r * SEQL + jb * 64 + c, &vt[buf][r * TP + c]);
    }
  };

  stage(0, 0);
  for (int jb = 0; jb < NJB; ++jb) {
    const int buf = jb & 1;
    __syncthreads();                       // all waves done reading buf^1
    if (jb + 1 < NJB) {
      stage(jb + 1, buf ^ 1);              // prefetch next tile via DMA
#ifdef HAVE_ASYNC_LDS
      __builtin_amdgcn_s_wait_asynccnt(8); // tile jb (issued earlier) complete
#endif
    } else {
#ifdef HAVE_ASYNC_LDS
      __builtin_amdgcn_s_wait_asynccnt(0);
#endif
    }
    __syncthreads();                       // tile jb visible to all waves

    // scores S = Q x K^T : 4 col tiles x 2 K-steps
    v8f s[4];
#pragma unroll
    for (int c = 0; c < 4; ++c) {
      v8f t = {};
#pragma unroll
      for (int ks = 0; ks < 2; ++ks) {
        const bf16* p = &kt[buf][(c * 16 + ln) * TP + ks * 32 + lh * 8];
        t = wmma_bf16(qf[ks], cat8(*(const v8bf*)p, *(const v8bf*)(p + 16)), t);
      }
      s[c] = t;
    }

    // online softmax: row (g + 8*half) lives on one 16-lane DPP row
#pragma unroll
    for (int g = 0; g < 8; ++g) {
      float mx = -3.0e38f;
#pragma unroll
      for (int c = 0; c < 4; ++c) { s[c][g] *= scale; mx = fmaxf(mx, s[c][g]); }
      mx = rowred_max(mx);
      const float mnew  = fmaxf(mrow[g], mx);
      const float alpha = __expf(mrow[g] - mnew);
      mrow[g] = mnew;
      float sum = 0.0f;
#pragma unroll
      for (int c = 0; c < 4; ++c) {
        const float e = __expf(s[c][g] - mnew);
        s[c][g] = e;
        sum += e;
      }
      sum = rowred_sum(sum);
      lrow[g] = lrow[g] * alpha + sum;
#pragma unroll
      for (int c = 0; c < 4; ++c) o[c][g] *= alpha;
    }

    // C-layout -> A-layout relayout of P through per-wave LDS strip
#pragma unroll
    for (int c = 0; c < 4; ++c)
#pragma unroll
      for (int g = 0; g < 8; ++g)
        pt[wave][(g + 8 * lh) * TP + c * 16 + ln] = (bf16)s[c][g];

    // O += P x V
#pragma unroll
    for (int ks = 0; ks < 2; ++ks) {
      const bf16* pp = pt[wave] + ln * TP + ks * 32 + lh * 8;
      const v16bf pa = cat8(*(const v8bf*)pp, *(const v8bf*)(pp + 16));
#pragma unroll
      for (int c = 0; c < 4; ++c) {
        const bf16* vp = &vt[buf][(c * 16 + ln) * TP + ks * 32 + lh * 8];
        o[c] = wmma_bf16(pa, cat8(*(const v8bf*)vp, *(const v8bf*)(vp + 16)), o[c]);
      }
    }
  }

  // finalize: O /= l, store ctx bf16 in [B, S, H*Dk]
#pragma unroll
  for (int g = 0; g < 8; ++g) {
    const float inv = 1.0f / lrow[g];
    const int srow = q0 + g + 8 * lh;
    bf16* dst = ctx + (size_t)(b * SEQL + srow) * DMODEL + h * DK;
#pragma unroll
    for (int c = 0; c < 4; ++c)
      dst[c * 16 + ln] = (bf16)(o[c][g] * inv);
  }
}

// ---------------------------------------------------------------------------
extern "C" void kernel_launch(void* const* d_in, const int* in_sizes, int n_in,
                              void* d_out, int out_size, void* d_ws, size_t ws_size,
                              hipStream_t stream) {
  (void)in_sizes; (void)n_in; (void)out_size; (void)ws_size;

  const float* Q  = (const float*)d_in[0];
  const float* K  = (const float*)d_in[1];
  const float* V  = (const float*)d_in[2];
  const float* Wq = (const float*)d_in[3];
  const float* bq = (const float*)d_in[4];
  const float* Wk = (const float*)d_in[5];
  const float* bk = (const float*)d_in[6];
  const float* Wv = (const float*)d_in[7];
  const float* bv = (const float*)d_in[8];
  const float* Wo = (const float*)d_in[9];
  const float* bo = (const float*)d_in[10];

  const size_t HS = (size_t)BATCH * NHEADS * SEQL * DK;  // 8Mi elems
  bf16* qh = (bf16*)d_ws;
  bf16* kh = qh + HS;
  bf16* vh = kh + HS;          // [B,H,Dk,S] transposed layout
  bf16* cx = vh + HS;          // total ws use = 64 MB

  const dim3 gg(DMODEL / BN, (BATCH * SEQL) / BM);  // 8 x 64 blocks

  gemm_dmodel<0><<<gg, 256, 0, stream>>>(Q, Wq, bq, qh);
  gemm_dmodel<0><<<gg, 256, 0, stream>>>(K, Wk, bk, kh);
  gemm_dmodel<2><<<gg, 256, 0, stream>>>(V, Wv, bv, vh);

  attn_fused<<<dim3(SEQL / 64, NHEADS, BATCH), 128, 0, stream>>>(qh, kh, vh, cx);

  gemm_dmodel<1><<<gg, 256, 0, stream>>>(cx, Wo, bo, d_out);
}
